// mha_69715909149527
// MI455X (gfx1250) — matrix operand order repaired and checked by reading, not compiled, not run on hardware
//
#include <hip/hip_runtime.h>
#include <hip/hip_bf16.h>

typedef _Float16 h16;
typedef __attribute__((ext_vector_type(16))) _Float16 v16h;
typedef __attribute__((ext_vector_type(8)))  _Float16 v8h;
typedef __attribute__((ext_vector_type(8)))  float    v8f;
typedef __attribute__((ext_vector_type(4)))  int      v4i;

#define BB 2
#define SS 2048
#define HH 16
#define DD 64
#define HD 1024
#define MM (BB*SS)   // 4096
#define CHK 64       // kv chunk in attention

#define AS1 __attribute__((address_space(1)))
#define AS3 __attribute__((address_space(3)))

#if __has_builtin(__builtin_amdgcn_global_load_async_to_lds_b128) && \
    __has_builtin(__builtin_amdgcn_s_wait_asynccnt)
#define HAVE_ASYNC 1
#endif

// one 16-byte global -> LDS copy (async DMA when available)
__device__ __forceinline__ void async_cp16(const h16* g, h16* l) {
#ifdef HAVE_ASYNC
    AS1 void* gv = (AS1 void*)const_cast<h16*>(g);   // addrspacecast generic->global
    AS3 void* lv = (AS3 void*)l;                     // addrspacecast generic->LDS
    __builtin_amdgcn_global_load_async_to_lds_b128((AS1 v4i*)gv, (AS3 v4i*)lv, 0, 0);
#else
    *(v8h*)l = *(const v8h*)g;   // global_load_b128 + ds_store_b128
#endif
}

template <int N>
__device__ __forceinline__ void wait_async() {
#ifdef HAVE_ASYNC
    __builtin_amdgcn_s_wait_asynccnt(N);
#endif
}

// Assemble a 16-half WMMA operand from two 16-byte chunks (two b128 loads).
__device__ __forceinline__ v16h mk_frag(const h16* p0, const h16* p1) {
    v8h a = *(const v8h*)p0;
    v8h b = *(const v8h*)p1;
    v16h r;
#pragma unroll
    for (int i = 0; i < 8; ++i) { r[i] = a[i]; r[i + 8] = b[i]; }
    return r;
}

__device__ __forceinline__ v8f wmma_f16(v16h a, v16h b, v8f c) {
    return __builtin_amdgcn_wmma_f32_16x16x32_f16(false, a, false, b,
                                                  (short)0, c, false, false);
}

// ---------------- elementwise f32 -> f16 convert -----------------------------
__global__ void k_cvt_f16(const float* __restrict__ x, h16* __restrict__ y, int n) {
    for (int i = blockIdx.x * blockDim.x + threadIdx.x; i < n;
         i += gridDim.x * blockDim.x)
        y[i] = (h16)x[i];
}

// --------------- transpose-convert weight: W[K][N] f32 -> Wt[N][K] f16 -------
__global__ void k_tr_f16(const float* __restrict__ W, h16* __restrict__ Wt) {
    __shared__ float t[32][33];
    int n0 = blockIdx.x * 32, k0 = blockIdx.y * 32;
    int tx = threadIdx.x, ty = threadIdx.y;                 // 32 x 8
#pragma unroll
    for (int i = 0; i < 32; i += 8)
        t[ty + i][tx] = W[(k0 + ty + i) * HD + n0 + tx];
    __syncthreads();
#pragma unroll
    for (int i = 0; i < 32; i += 8)
        Wt[(n0 + ty + i) * HD + k0 + tx] = (h16)t[tx][ty + i];
}

// --------------- shared GEMM core: 64(M) x 64(N) per block, 4 waves ----------
// A: row-major f16 [*, HD]; Wt: column-major weights [N][K] f16.
// B tile (64 n x 32 k) is double-buffered in LDS via async global->LDS copies.
__device__ __forceinline__ void gemm_block(const h16* __restrict__ Arow,
                                           const h16* __restrict__ WtN0,
                                           h16 (*Bt)[64 * 32],
                                           int tid, int lq, int hs,
                                           v8f acc[4]) {
    // issue: 256 b128 chunks / 128 threads = 2 per thread
#define GEMM_ISSUE(buf, kk)                                                     \
    {                                                                           \
        _Pragma("unroll")                                                       \
        for (int j = 0; j < 2; ++j) {                                           \
            int c = tid + 128 * j, row = c >> 2, qo = c & 3;                    \
            async_cp16(WtN0 + (size_t)row * HD + (kk) + qo * 8,                 \
                       &Bt[buf][row * 32 + qo * 8]);                            \
        }                                                                       \
    }
    GEMM_ISSUE(0, 0)
    const h16* ap0 = Arow + 16 * hs;
    v16h a = mk_frag(ap0, ap0 + 8);
    for (int it = 0; it < HD / 32; ++it) {
        const int cur = it & 1, kk = it * 32;
        if (it + 1 < HD / 32) { GEMM_ISSUE(cur ^ 1, kk + 32) wait_async<2>(); }
        else                  { wait_async<0>(); }
        __syncthreads();
        v16h an;
        if (it + 1 < HD / 32) {                 // overlap next-A global load
            const h16* ap = Arow + kk + 32 + 16 * hs;
            an = mk_frag(ap, ap + 8);
        }
#pragma unroll
        for (int t = 0; t < 4; ++t) {
            const h16* bp = &Bt[cur][(t * 16 + lq) * 32 + 16 * hs];
            acc[t] = wmma_f16(a, mk_frag(bp, bp + 8), acc[t]);   // ds_load_b128 x2
        }
        __syncthreads();
        a = an;
    }
#undef GEMM_ISSUE
}

// --------------- QKV projection GEMM: Y = x @ W + b --------------------------
__global__ void __launch_bounds__(128)
k_qkv(const h16* __restrict__ xh, const h16* __restrict__ Wt3,
      const float* __restrict__ bq, const float* __restrict__ bk,
      const float* __restrict__ bv,
      h16* __restrict__ Qo, h16* __restrict__ Ko, h16* __restrict__ Vto) {
    __shared__ h16 Bt[2][64 * 32];
    const int tid = threadIdx.x, wave = tid >> 5, lane = tid & 31;
    const int lq = lane & 15, hs = lane >> 4;
    const int m0 = blockIdx.x * 64 + wave * 16;
    const int n0 = blockIdx.y * 64;
    const int p  = blockIdx.z;                // 0=q 1=k 2=v
    const h16* Wt = Wt3 + (size_t)p * HD * HD;
    const float* bias = (p == 0) ? bq : (p == 1) ? bk : bv;

    v8f acc[4] = {};
    gemm_block(xh + (size_t)(m0 + lq) * HD, Wt + (size_t)n0 * HD,
               Bt, tid, lq, hs, acc);
#pragma unroll
    for (int t = 0; t < 4; ++t) {
        const int n = n0 + t * 16 + lq;
        const float bsv = bias[n];
        const int h = n >> 6, d = n & 63;
#pragma unroll
        for (int r = 0; r < 8; ++r) {
            const int m = m0 + r + 8 * hs;
            const int b = m >> 11, s = m & 2047;
            const float v = acc[t][r] + bsv;
            if (p == 0)      Qo [(((size_t)(b * HH + h) * SS + s) * DD) + d] = (h16)v;
            else if (p == 1) Ko [(((size_t)(b * HH + h) * SS + s) * DD) + d] = (h16)v;
            else             Vto[(((size_t)(b * HH + h) * DD + d) * SS) + s] = (h16)v;
        }
    }
}

// --------------- flash attention: 4 waves/block share one (b,h) K/V stream ---
__global__ void __launch_bounds__(128)
k_attn(const h16* __restrict__ Q, const h16* __restrict__ K,
       const h16* __restrict__ Vt, h16* __restrict__ Oh) {
    __shared__ h16 Kt[2][CHK * 64];    // [kv][d]  8 KB each
    __shared__ h16 Vtt[2][64 * CHK];   // [d][kv]  8 KB each
    __shared__ h16 Pl[4][16 * CHK];    // per-wave P tile (16 x 64)
    const int tid = threadIdx.x, wave = tid >> 5, lane = tid & 31;
    const int lq = lane & 15, hs = lane >> 4;
    const int qg = blockIdx.x & 31, h = (blockIdx.x >> 5) & 15, b = blockIdx.x >> 9;

    const h16* Qb = Q  + (size_t)(b * HH + h) * SS * DD;
    const h16* Kb = K  + (size_t)(b * HH + h) * SS * DD;
    const h16* Vb = Vt + (size_t)(b * HH + h) * DD * SS;
    const int q0 = (qg * 4 + wave) * 16;

    v16h qa[2];
#pragma unroll
    for (int kk = 0; kk < 2; ++kk) {
        const h16* p = Qb + (size_t)(q0 + lq) * DD + kk * 32 + 16 * hs;
        qa[kk] = mk_frag(p, p + 8);
    }

    v8f o[4] = {};
    float mrow[8], lrow[8];
#pragma unroll
    for (int r = 0; r < 8; ++r) { mrow[r] = -__builtin_huge_valf(); lrow[r] = 0.f; }
    h16* Pw = &Pl[wave][0];

    // 1024 b128 chunks / 128 threads = 8 per thread (4 for K tile, 4 for V tile)
#define ATTN_ISSUE(buf, kv)                                                     \
    {                                                                           \
        _Pragma("unroll")                                                       \
        for (int j = 0; j < 4; ++j) {                                           \
            int c = tid + 128 * j, row = c >> 3, qo = c & 7;                    \
            async_cp16(Kb + (size_t)((kv) + row) * DD + qo * 8,                 \
                       &Kt[buf][row * 64 + qo * 8]);                            \
        }                                                                       \
        _Pragma("unroll")                                                       \
        for (int j = 0; j < 4; ++j) {                                           \
            int c = tid + 128 * j, row = c >> 3, qo = c & 7;                    \
            async_cp16(Vb + (size_t)row * SS + (kv) + qo * 8,                   \
                       &Vtt[buf][row * 64 + qo * 8]);                           \
        }                                                                       \
    }
    ATTN_ISSUE(0, 0)
    for (int it = 0; it < SS / CHK; ++it) {
        const int cur = it & 1;
        if (it + 1 < SS / CHK) { ATTN_ISSUE(cur ^ 1, (it + 1) * CHK) wait_async<8>(); }
        else                   { wait_async<0>(); }
        __syncthreads();
        // scores: Q[16x64] @ K^T[64x64] -> 4 C frags
        v8f s[4];
#pragma unroll
        for (int t = 0; t < 4; ++t) {
            v8f acc = {};
#pragma unroll
            for (int kk = 0; kk < 2; ++kk) {
                const h16* bp = &Kt[cur][(t * 16 + lq) * 64 + kk * 32 + 16 * hs];
                acc = wmma_f16(qa[kk], mk_frag(bp, bp + 8), acc);
            }
            s[t] = acc;
        }
        // online softmax, scale = 0.125
        float tmax[8];
#pragma unroll
        for (int r = 0; r < 8; ++r) {
            float a0 = s[0][r] * 0.125f, a1 = s[1][r] * 0.125f;
            float a2 = s[2][r] * 0.125f, a3 = s[3][r] * 0.125f;
            s[0][r] = a0; s[1][r] = a1; s[2][r] = a2; s[3][r] = a3;
            tmax[r] = fmaxf(fmaxf(a0, a1), fmaxf(a2, a3));
        }
#pragma unroll
        for (int msk = 1; msk <= 8; msk <<= 1)
#pragma unroll
            for (int r = 0; r < 8; ++r)
                tmax[r] = fmaxf(tmax[r], __shfl_xor(tmax[r], msk, 32));
        float alpha[8], rsum[8];
#pragma unroll
        for (int r = 0; r < 8; ++r) {
            float mn = fmaxf(mrow[r], tmax[r]);
            alpha[r] = __expf(mrow[r] - mn);
            mrow[r] = mn;
            float e0 = __expf(s[0][r] - mn), e1 = __expf(s[1][r] - mn);
            float e2 = __expf(s[2][r] - mn), e3 = __expf(s[3][r] - mn);
            s[0][r] = e0; s[1][r] = e1; s[2][r] = e2; s[3][r] = e3;
            rsum[r] = (e0 + e1) + (e2 + e3);
        }
#pragma unroll
        for (int msk = 1; msk <= 8; msk <<= 1)
#pragma unroll
            for (int r = 0; r < 8; ++r)
                rsum[r] += __shfl_xor(rsum[r], msk, 32);
#pragma unroll
        for (int r = 0; r < 8; ++r) lrow[r] = lrow[r] * alpha[r] + rsum[r];
#pragma unroll
        for (int t = 0; t < 4; ++t)
#pragma unroll
            for (int r = 0; r < 8; ++r) o[t][r] *= alpha[r];
        // C-layout -> A-layout via per-wave LDS bounce
#pragma unroll
        for (int t = 0; t < 4; ++t)
#pragma unroll
            for (int r = 0; r < 8; ++r)
                Pw[(r + 8 * hs) * CHK + t * 16 + lq] = (h16)s[t][r];
        v16h pa[2];
#pragma unroll
        for (int ks = 0; ks < 2; ++ks) {
            const h16* pp = Pw + lq * CHK + ks * 32 + 16 * hs;
            pa[ks] = mk_frag(pp, pp + 8);            // ds_load_b128 x2
        }
        // O += P[16x64] @ V[64x64]
#pragma unroll
        for (int t = 0; t < 4; ++t)
#pragma unroll
            for (int ks = 0; ks < 2; ++ks) {
                const h16* vp = &Vtt[cur][(t * 16 + lq) * 64 + ks * 32 + 16 * hs];
                o[t] = wmma_f16(pa[ks], mk_frag(vp, vp + 8), o[t]);
            }
        __syncthreads();
    }
#undef ATTN_ISSUE
    float inv[8];
#pragma unroll
    for (int r = 0; r < 8; ++r) inv[r] = 1.0f / lrow[r];
    h16* Ob = Oh + ((size_t)(b * SS + q0)) * HD + h * DD;
#pragma unroll
    for (int t = 0; t < 4; ++t)
#pragma unroll
        for (int r = 0; r < 8; ++r)
            Ob[(size_t)(r + 8 * hs) * HD + t * 16 + lq] = (h16)(o[t][r] * inv[r]);
}

// --------------- output projection: out = O @ Wo + bo (fp32 out) -------------
__global__ void __launch_bounds__(128)
k_oproj(const h16* __restrict__ Oh, const h16* __restrict__ Wto,
        const float* __restrict__ bo, float* __restrict__ out) {
    __shared__ h16 Bt[2][64 * 32];
    const int tid = threadIdx.x, wave = tid >> 5, lane = tid & 31;
    const int lq = lane & 15, hs = lane >> 4;
    const int m0 = blockIdx.x * 64 + wave * 16;
    const int n0 = blockIdx.y * 64;

    v8f acc[4] = {};
    gemm_block(Oh + (size_t)(m0 + lq) * HD, Wto + (size_t)n0 * HD,
               Bt, tid, lq, hs, acc);
#pragma unroll
    for (int t = 0; t < 4; ++t) {
        const int n = n0 + t * 16 + lq;
        const float bsv = bo[n];
#pragma unroll
        for (int r = 0; r < 8; ++r)
            out[(size_t)(m0 + r + 8 * hs) * HD + n] = acc[t][r] + bsv;
    }
}

extern "C" void kernel_launch(void* const* d_in, const int* in_sizes, int n_in,
                              void* d_out, int out_size, void* d_ws, size_t ws_size,
                              hipStream_t stream) {
    const float* x  = (const float*)d_in[0];
    const float* Wq = (const float*)d_in[1];
    const float* bq = (const float*)d_in[2];
    const float* Wk = (const float*)d_in[3];
    const float* bk = (const float*)d_in[4];
    const float* Wv = (const float*)d_in[5];
    const float* bv = (const float*)d_in[6];
    const float* Wo = (const float*)d_in[7];
    const float* bo = (const float*)d_in[8];
    float* out = (float*)d_out;

    // workspace layout (f16 elements), total 24M halves = 48 MB
    h16* xh  = (h16*)d_ws;                       // [4096][1024]
    h16* Wt3 = xh  + (size_t)MM * HD;            // 3 x [1024][1024] col-major
    h16* Wto = Wt3 + (size_t)3 * HD * HD;        // [1024][1024] col-major
    h16* Qb  = Wto + (size_t)HD * HD;            // [B,H,S,D]
    h16* Kb  = Qb  + (size_t)BB * HH * SS * DD;  // [B,H,S,D]
    h16* Vtb = Kb  + (size_t)BB * HH * SS * DD;  // [B,H,D,S]
    h16* Ohb = Vtb + (size_t)BB * HH * SS * DD;  // [4096][1024]

    k_cvt_f16<<<4096, 256, 0, stream>>>(x, xh, MM * HD);
    dim3 trb(32, 8), trg(32, 32);
    k_tr_f16<<<trg, trb, 0, stream>>>(Wq, Wt3);
    k_tr_f16<<<trg, trb, 0, stream>>>(Wk, Wt3 + (size_t)HD * HD);
    k_tr_f16<<<trg, trb, 0, stream>>>(Wv, Wt3 + (size_t)2 * HD * HD);
    k_tr_f16<<<trg, trb, 0, stream>>>(Wo, Wto);

    k_qkv<<<dim3(64, 16, 3), 128, 0, stream>>>(xh, Wt3, bq, bk, bv, Qb, Kb, Vtb);
    k_attn<<<1024, 128, 0, stream>>>(Qb, Kb, Vtb, Ohb);
    k_oproj<<<dim3(64, 16), 128, 0, stream>>>(Ohb, Wto, bo, out);
}
